// SpeechOdeTCNNModel_54236847013966
// MI455X (gfx1250) — matrix operand order repaired and checked by reading, not compiled, not run on hardware
//
#include <hip/hip_runtime.h>
#include <hip/hip_bf16.h>

// ---------------------------------------------------------------------------
// Speech ODE TCNN forward for MI455X (gfx1250, wave32, WMMA).
// - Convs = bf16 WMMA GEMMs (v_wmma_f32_16x16x32_bf16), f32 accumulate.
// - bf16 activations channel-last [b][t][c] -> conv staging is contiguous
//   global_load_async_to_lds_b128 (ASYNCcnt) into LDS.
// - LDS row stride = C+8 (272B): 16B-aligned async stores, 16B-aligned
//   ds_load_b128 B-fragment gathers (2-way conflicts max).
// - Conv kernel templated on shape: compile-time K-loop, immediate offsets.
// - BN uses batch stats => global reductions between convs (kernel pipeline).
// Working set ~200MB, essentially L2-resident (192MB L2, 23.3TB/s HBM).
// ---------------------------------------------------------------------------

typedef __attribute__((ext_vector_type(16))) __bf16 v16bf;
typedef __attribute__((ext_vector_type(8)))  float  v8f;
typedef __attribute__((ext_vector_type(8)))  int    v8i;

#define TMAX   500
#define CMAPS  128
#define BATCH  128
#define NLAB   35
#define TW     64     // time positions per conv block (4 N-tiles of 16)

__device__ __forceinline__ unsigned short f2bf(float f) {
  unsigned u = __builtin_bit_cast(unsigned, f);
  unsigned r = 0x7FFFu + ((u >> 16) & 1u);   // round-to-nearest-even
  return (unsigned short)((u + r) >> 16);
}

// ---------------------------------------------------------------------------
// Pack conv weights [Cout][CinReal][taps] (f32) into WMMA A-fragment order:
// wp[((m_tile*KC + kc)*32 + lane)*8 + v] = bf16 pair (K,K+1) for row M=lane%16
// per CDNA5 16-bit A 16x32 layout: K_lo(v,half) = (v>=4)*16 + half*8 + 2*(v&3)
// K ordered tap-major, channel-minor, Cin padded to CinPad.
// ---------------------------------------------------------------------------
__global__ void packw_kernel(const float* __restrict__ w, unsigned* __restrict__ wp,
                             int CinReal, int CinPad, int taps) {
  int KC   = taps * (CinPad >> 5);
  int blk  = blockIdx.x;
  int mt   = blk / KC;
  int kc   = blk - mt * KC;
  int tid  = threadIdx.x;
  int lane = tid >> 3, v = tid & 7;
  int M = lane & 15, half = lane >> 4;
  int klo = ((v >= 4) ? 16 : 0) + half * 8 + 2 * (v & 3);
  int Kg  = kc * 32 + klo;
  int tap = Kg / CinPad;
  int ci  = Kg - tap * CinPad;
  int row = mt * 16 + M;
  float a = 0.f, b = 0.f;
  if (ci < CinReal)     a = w[((size_t)row * CinReal + ci) * taps + tap];
  if (ci + 1 < CinReal) b = w[((size_t)row * CinReal + ci + 1) * taps + tap];
  wp[((size_t)blk * 32 + lane) * 8 + v] = ((unsigned)f2bf(b) << 16) | (unsigned)f2bf(a);
}

// x [b][80][T] f32 -> xbf [b][t][80] bf16 (LDS tile transpose, both sides
// coalesced). grid = (ceil(T/32), B), 256 threads.
__global__ __launch_bounds__(256) void x2bf_kernel(const float* __restrict__ x,
                                                   unsigned short* __restrict__ o) {
  __shared__ unsigned short sT[32 * 88];
  int b = blockIdx.y, t0 = blockIdx.x * 32;
  int lane = threadIdx.x & 31, wave = threadIdx.x >> 5;
  int t = t0 + lane;
  for (int c0 = 0; c0 < 80; c0 += 8) {
    int c = c0 + wave;
    float v = 0.f;
    if (t < TMAX) v = x[((size_t)b * 80 + c) * TMAX + t];
    sT[lane * 88 + c] = f2bf(v);
  }
  __syncthreads();
  for (int e = threadIdx.x; e < 32 * 80; e += 256) {
    int tl = e / 80, c = e - tl * 80;
    int tt = t0 + tl;
    if (tt < TMAX) o[((size_t)b * TMAX + tt) * 80 + c] = sT[tl * 88 + c];
  }
}

// ---------------------------------------------------------------------------
// 1D conv as WMMA GEMM, templated on shape (compile-time K-loop, immediate
// LDS/global offsets). Block = 256 threads (8 wave32); grid = (T/TW, B).
// Wave w owns output channels [16w,16w+16); block covers 64 time positions.
// ---------------------------------------------------------------------------
template <int CINREAL, int CINPAD, int TAPS, int PADH>
__global__ __launch_bounds__(256) void conv_wmma_kernel(
    const unsigned short* __restrict__ in, const unsigned* __restrict__ wpack,
    float* __restrict__ out) {
  constexpr int STRIDEP = CINPAD + 8;        // ushort row stride (16B multiple*2)
  constexpr int WIDTH   = TW + TAPS - 1;
  constexpr int CGROUPS = CINPAD / 32;
  constexpr int KC      = TAPS * CGROUPS;
  constexpr int NGRAN   = CINPAD / 8;        // 16B granules per row
  constexpr int GREAL   = CINREAL / 8;       // granules holding real data
  __shared__ unsigned short lds[WIDTH * STRIDEP];
  const int T = TMAX;
  int b  = blockIdx.y;
  int t0 = blockIdx.x * TW;
  int tid = threadIdx.x;

  // Async-stage activations: global [b][t][ci] -> LDS [t_local][ci].
  for (int e = tid; e < WIDTH * NGRAN; e += 256) {
    int tl = e / NGRAN;
    int g  = e - tl * NGRAN;
    int t  = t0 - PADH + tl;
    unsigned short* dst = &lds[tl * STRIDEP + g * 8];
    if (t >= 0 && t < T && g < GREAL) {
      unsigned lv = (unsigned)(size_t)dst;   // LDS aperture: low 32b = LDS addr
      const unsigned short* src = in + ((size_t)b * T + t) * CINREAL + g * 8;
      asm volatile("global_load_async_to_lds_b128 %0, %1, off"
                   :: "v"(lv), "v"((unsigned long long)(size_t)src)
                   : "memory");
    } else {
      *(uint4*)dst = make_uint4(0u, 0u, 0u, 0u);
    }
  }
  asm volatile("s_wait_asynccnt 0x0" ::: "memory");
  __syncthreads();

  int wave = tid >> 5, lane = tid & 31;
  int lanelo = lane & 15, khalf = lane >> 4;
  const uint4* ldsv = (const uint4*)lds;
  int laneoff = (lanelo * STRIDEP + khalf * 8) >> 3;    // uint4 units (exact)

  v8f acc0 = {}, acc1 = {}, acc2 = {}, acc3 = {};
  const uint4* wp = (const uint4*)wpack + ((size_t)wave * KC * 32 + lane) * 2;

#pragma unroll
  for (int tap = 0; tap < TAPS; ++tap) {
#pragma unroll
    for (int cg = 0; cg < CGROUPS; ++cg) {
      uint4 qa = wp[0];
      uint4 qb = wp[1];
      wp += 64;                       // 32 lanes * 2 uint4 per kc entry
      v8i ai;
      ai[0] = (int)qa.x; ai[1] = (int)qa.y; ai[2] = (int)qa.z; ai[3] = (int)qa.w;
      ai[4] = (int)qb.x; ai[5] = (int)qb.y; ai[6] = (int)qb.z; ai[7] = (int)qb.w;
      v16bf a = __builtin_bit_cast(v16bf, ai);
      int kb = ((tap * STRIDEP + cg * 32) >> 3) + laneoff;   // uint4 units
#pragma unroll
      for (int nt = 0; nt < 4; ++nt) {
        int q = kb + nt * 2 * STRIDEP;        // uint4 units per 16-t step
        uint4 b0 = ldsv[q + 0];
        uint4 b1 = ldsv[q + 2];
        v8i bi;
        bi[0] = (int)b0.x; bi[1] = (int)b0.y; bi[2] = (int)b0.z; bi[3] = (int)b0.w;
        bi[4] = (int)b1.x; bi[5] = (int)b1.y; bi[6] = (int)b1.z; bi[7] = (int)b1.w;
        v16bf bm = __builtin_bit_cast(v16bf, bi);
        v8f cc = (nt == 0) ? acc0 : (nt == 1) ? acc1 : (nt == 2) ? acc2 : acc3;
        cc = __builtin_amdgcn_wmma_f32_16x16x32_bf16(false, a, false, bm,
                                                     (short)0, cc, false, false);
        if (nt == 0) acc0 = cc; else if (nt == 1) acc1 = cc;
        else if (nt == 2) acc2 = cc; else acc3 = cc;
      }
    }
  }

  // Store C tiles: lane holds (M = 8*khalf + r, N = lanelo), f32 out [b][c][t].
  size_t ob = ((size_t)b * CMAPS + wave * 16) * T;
#pragma unroll
  for (int nt = 0; nt < 4; ++nt) {
    v8f cc = (nt == 0) ? acc0 : (nt == 1) ? acc1 : (nt == 2) ? acc2 : acc3;
    int t = t0 + nt * 16 + lanelo;
    if (t < T) {
#pragma unroll
      for (int r = 0; r < 8; ++r) {
        int M = khalf * 8 + r;
        out[ob + (size_t)M * T + t] = cc[r];
      }
    }
  }
}

// Per-channel batch-norm statistics over (B,T): mean and rsqrt(var+eps).
__global__ void bnstats_kernel(const float* __restrict__ x,
                               float* __restrict__ mean, float* __restrict__ scal) {
  const int BT = BATCH * TMAX;
  int c = blockIdx.x;
  float s1 = 0.f, s2 = 0.f;
  for (int i = threadIdx.x; i < BT; i += 256) {
    int b = i / TMAX, t = i - b * TMAX;
    float v = x[((size_t)b * CMAPS + c) * TMAX + t];
    s1 += v; s2 += v * v;
  }
  __shared__ float r1[8], r2[8];
  for (int off = 16; off; off >>= 1) {
    s1 += __shfl_down(s1, off, 32);
    s2 += __shfl_down(s2, off, 32);
  }
  int wave = threadIdx.x >> 5, lane = threadIdx.x & 31;
  if (lane == 0) { r1[wave] = s1; r2[wave] = s2; }
  __syncthreads();
  if (threadIdx.x == 0) {
    float a = 0.f, q = 0.f;
    for (int w = 0; w < 8; ++w) { a += r1[w]; q += r2[w]; }
    float m = a / BT;
    float var = q / BT - m * m;
    mean[c] = m;
    scal[c] = rsqrtf(var + 1e-5f);
  }
}

// relu((x-m)*s): f32 side stays [b][c][t] (coalesced), bf16 side transposed
// through LDS to channel-last [b][t][c]. grid = (ceil(T/32), B).
__global__ __launch_bounds__(256) void bnrelu_kernel(
    const float* __restrict__ x, const float* __restrict__ mean,
    const float* __restrict__ scal,
    unsigned short* __restrict__ obf, float* __restrict__ of32) {
  __shared__ unsigned short sT[32 * 136];
  int b = blockIdx.y, t0 = blockIdx.x * 32;
  int lane = threadIdx.x & 31, wave = threadIdx.x >> 5;
  int t = t0 + lane;
  for (int c0 = 0; c0 < CMAPS; c0 += 8) {
    int c = c0 + wave;
    float v = 0.f;
    if (t < TMAX) {
      size_t idx = ((size_t)b * CMAPS + c) * TMAX + t;
      v = fmaxf((x[idx] - mean[c]) * scal[c], 0.f);
      if (of32) of32[idx] = v;
    }
    sT[lane * 136 + c] = f2bf(v);
  }
  __syncthreads();
  for (int e = threadIdx.x; e < 32 * CMAPS; e += 256) {
    int tl = e >> 7, c = e & (CMAPS - 1);
    int tt = t0 + tl;
    if (tt < TMAX) obf[((size_t)b * TMAX + tt) * CMAPS + c] = sT[tl * 136 + c];
  }
}

// k = relu(bn(t2) + relu(bn(t3))); acc (+)= wk*k; next eval point -> bf16
// (channel-last, via LDS transpose).
// mode 0: no abf; mode 1: abf = bf16(z + cnext*k); mode 2: abf = bf16(acc_new)
__global__ __launch_bounds__(256) void combine_kernel(
    const float* __restrict__ t2, const float* __restrict__ t3,
    const float* __restrict__ stats, const float* __restrict__ zcur,
    float* __restrict__ acc, unsigned short* __restrict__ abf,
    float wk, float cnext, int mode, int init) {
  __shared__ unsigned short sT[32 * 136];
  int b = blockIdx.y, t0 = blockIdx.x * 32;
  int lane = threadIdx.x & 31, wave = threadIdx.x >> 5;
  int t = t0 + lane;
  for (int c0 = 0; c0 < CMAPS; c0 += 8) {
    int c = c0 + wave;
    float val = 0.f;
    if (t < TMAX) {
      size_t idx = ((size_t)b * CMAPS + c) * TMAX + t;
      float m2 = stats[256 + c], s2 = stats[384 + c];
      float m3 = stats[512 + c], s3 = stats[640 + c];
      float o3 = fmaxf((t3[idx] - m3) * s3, 0.f);
      float k  = fmaxf((t2[idx] - m2) * s2 + o3, 0.f);
      float zv = zcur[idx];
      float an = (init ? zv : acc[idx]) + wk * k;
      acc[idx] = an;
      val = (mode == 1) ? (zv + cnext * k) : an;
    }
    sT[lane * 136 + c] = f2bf(val);
  }
  if (mode == 0) return;    // uniform across block
  __syncthreads();
  for (int e = threadIdx.x; e < 32 * CMAPS; e += 256) {
    int tl = e >> 7, c = e & (CMAPS - 1);
    int tt = t0 + tl;
    if (tt < TMAX) abf[((size_t)b * TMAX + tt) * CMAPS + c] = sT[tl * 136 + c];
  }
}

// Global average pool over T + linear head. grid = (B), 256 threads.
__global__ void head_kernel(const float* __restrict__ z, const float* __restrict__ ow,
                            const float* __restrict__ ob, float* __restrict__ out) {
  int b = blockIdx.x;
  __shared__ float feat[CMAPS];
  int wave = threadIdx.x >> 5, lane = threadIdx.x & 31;
  for (int c = wave; c < CMAPS; c += 8) {
    const float* p = z + ((size_t)b * CMAPS + c) * TMAX;
    float s = 0.f;
    for (int t = lane; t < TMAX; t += 32) s += p[t];
    for (int off = 16; off; off >>= 1) s += __shfl_down(s, off, 32);
    if (lane == 0) feat[c] = s * (1.f / TMAX);
  }
  __syncthreads();
  if (threadIdx.x < NLAB) {
    float s = ob[threadIdx.x];
    for (int c = 0; c < CMAPS; ++c) s += feat[c] * ow[threadIdx.x * CMAPS + c];
    out[b * NLAB + threadIdx.x] = s;
  }
}

extern "C" void kernel_launch(void* const* d_in, const int* in_sizes, int n_in,
                              void* d_out, int out_size, void* d_ws, size_t ws_size,
                              hipStream_t stream) {
  const float* x  = (const float*)d_in[0];
  const float* w0 = (const float*)d_in[1];
  const float* w1 = (const float*)d_in[2];
  const float* w2 = (const float*)d_in[3];
  const float* w3 = (const float*)d_in[4];
  const float* ow = (const float*)d_in[5];
  const float* ob = (const float*)d_in[6];
  float* out = (float*)d_out;

  char* ws = (char*)d_ws;
  const size_t S = (size_t)BATCH * CMAPS * TMAX * 4;       // 32,768,000 B
  float* zA = (float*)(ws + 0 * S);
  float* zB = (float*)(ws + 1 * S);
  float* t1 = (float*)(ws + 2 * S);
  float* t2 = (float*)(ws + 3 * S);
  float* t3 = (float*)(ws + 4 * S);
  unsigned short* abf = (unsigned short*)(ws + 5 * S);              // [b][t][128] bf16
  unsigned short* xbf = (unsigned short*)(ws + 5 * S + (S >> 1));   // [b][t][80]  bf16
  char* wb = ws + 5 * S + (S >> 1) + (size_t)BATCH * 80 * TMAX * 2;
  unsigned* w1p = (unsigned*)(wb);
  unsigned* w2p = (unsigned*)(wb + 294912);
  unsigned* w0p = (unsigned*)(wb + 2 * 294912);
  unsigned* w3p = (unsigned*)(wb + 2 * 294912 + 73728);
  float* stats  = (float*)(wb + 2 * 294912 + 73728 + 32768);  // 6x128 floats used

  dim3 blk(256);
  dim3 cg(8, BATCH);              // conv grid: 8 time tiles x 128 batches
  dim3 tg(16, BATCH);             // transpose/elementwise grid: 16 t-chunks x B

  // One-time (per call) prep: weight packing + x -> channel-last bf16.
  packw_kernel<<<dim3(8 * 9),  blk, 0, stream>>>(w0, w0p, 80, 96, 3);
  packw_kernel<<<dim3(8 * 36), blk, 0, stream>>>(w1, w1p, 128, 128, 9);
  packw_kernel<<<dim3(8 * 36), blk, 0, stream>>>(w2, w2p, 128, 128, 9);
  packw_kernel<<<dim3(8 * 4),  blk, 0, stream>>>(w3, w3p, 128, 128, 1);
  x2bf_kernel<<<tg, blk, 0, stream>>>(x, xbf);

  // h0 = relu(bn(conv0(x))) -> zA (f32) and abf (bf16 eval point for k1)
  conv_wmma_kernel<80, 96, 3, 1><<<cg, blk, 0, stream>>>(xbf, w0p, t1);
  bnstats_kernel<<<dim3(CMAPS), blk, 0, stream>>>(t1, stats + 0, stats + 128);
  bnrelu_kernel<<<tg, blk, 0, stream>>>(t1, stats + 0, stats + 128, abf, zA);

  const float dt = 0.25f;
  const float wk[4] = { dt / 6.f, dt / 3.f, dt / 3.f, dt / 6.f };
  const float cn[4] = { 0.5f * dt, 0.5f * dt, dt, 0.f };

  for (int s = 0; s < 4; ++s) {
    float* zcur = (s & 1) ? zB : zA;
    float* acc  = (s & 1) ? zA : zB;
    for (int e = 0; e < 4; ++e) {
      // odefunc at eval point (bf16 channel-last in abf)
      conv_wmma_kernel<128, 128, 9, 4><<<cg, blk, 0, stream>>>(abf, w1p, t1);
      conv_wmma_kernel<128, 128, 1, 0><<<cg, blk, 0, stream>>>(abf, w3p, t3);
      bnstats_kernel<<<dim3(CMAPS), blk, 0, stream>>>(t1, stats + 0, stats + 128);
      bnstats_kernel<<<dim3(CMAPS), blk, 0, stream>>>(t3, stats + 512, stats + 640);
      bnrelu_kernel<<<tg, blk, 0, stream>>>(t1, stats + 0, stats + 128, abf, nullptr);
      conv_wmma_kernel<128, 128, 9, 4><<<cg, blk, 0, stream>>>(abf, w2p, t2);
      bnstats_kernel<<<dim3(CMAPS), blk, 0, stream>>>(t2, stats + 256, stats + 384);
      int mode = (e < 3) ? 1 : ((s < 3) ? 2 : 0);
      combine_kernel<<<tg, blk, 0, stream>>>(t2, t3, stats, zcur, acc, abf,
                                             wk[e], cn[e], mode, e == 0);
    }
  }
  // Final state is in zA after step 3 (acc of s=3).
  head_kernel<<<dim3(BATCH), blk, 0, stream>>>(zA, ow, ob, out);
}
